// ConvBlock_T40_35570919145966
// MI455X (gfx1250) — 8-bit matrix operand order repaired and checked by reading, compiled, not run on hardware
//
#include <hip/hip_runtime.h>
#include <hip/hip_bf16.h>
#include <stdint.h>


#define NB     32
#define CIN    128
#define COUT   256
#define HH     56
#define WW     56
#define HW     (HH * WW)
#define NPIX   (NB * HW)
#define NTAPS  9
#define HP     58

#define CLIP_MAX  6.0f
#define QA_SCALE  (255.0f / 6.0f)
#define QA_INV    (6.0f / 255.0f)
#define BN_EPS    1e-5f

typedef int v8i __attribute__((ext_vector_type(8)));
typedef int v4i_vs __attribute__((vector_size(16)));
typedef __attribute__((address_space(1))) v4i_vs* as1_v4i_p;
typedef __attribute__((address_space(3))) v4i_vs* as3_v4i_p;

__device__ __forceinline__ void async_copy16(const uint32_t* g, uint32_t* l) {
#if __has_builtin(__builtin_amdgcn_global_load_async_to_lds_b128)
    __builtin_amdgcn_global_load_async_to_lds_b128(
        (as1_v4i_p)(uintptr_t)g,
        (as3_v4i_p)(uint32_t)(uintptr_t)l,
        0, 0);
#else
    uint4 v = *(const uint4*)g;
    *(uint4*)l = v;
#endif
}

__device__ __forceinline__ void wait_async() {
#if __has_builtin(__builtin_amdgcn_s_wait_asynccnt)
    __builtin_amdgcn_s_wait_asynccnt(0);
#elif defined(__gfx1250__)
    asm volatile("s_wait_asynccnt 0" ::: "memory");
#endif
}

__global__ void k_init_smax(unsigned* smax) {
    if (threadIdx.x == 0 && blockIdx.x == 0) *smax = 0u;
}

__global__ __launch_bounds__(256)
void k_absmax(const float* __restrict__ w, unsigned* __restrict__ smax, int n) {
    __shared__ float red[256];
    float m = 0.0f;
    for (int i = blockIdx.x * blockDim.x + threadIdx.x; i < n;
         i += gridDim.x * blockDim.x)
        m = fmaxf(m, fabsf(w[i]));
    red[threadIdx.x] = m;
    __syncthreads();
    for (int s = 128; s > 0; s >>= 1) {
        if ((int)threadIdx.x < s)
            red[threadIdx.x] = fmaxf(red[threadIdx.x], red[threadIdx.x + s]);
        __syncthreads();
    }
    if (threadIdx.x == 0) atomicMax(smax, __float_as_uint(red[0]));
}

__global__ __launch_bounds__(256)
void k_quant_w(const float* __restrict__ w, const unsigned* __restrict__ smax,
               signed char* __restrict__ wq) {
    int gid = blockIdx.x * 256 + threadIdx.x;
    if (gid >= NTAPS * COUT * CIN) return;
    int ci = gid & (CIN - 1);
    int o  = (gid >> 7) & (COUT - 1);
    int t  = gid >> 15;
    int ky = t / 3, kx = t - 3 * ky;
    float s = fmaxf(__uint_as_float(*smax), 1e-8f);
    float v = w[(((size_t)o * CIN + ci) * 3 + ky) * 3 + kx];
    int q = (int)rintf(v * (127.0f / s));
    q = q < -127 ? -127 : (q > 127 ? 127 : q);
    wq[gid] = (signed char)q;
}

__global__ __launch_bounds__(256)
void k_zero_halo(uint4* __restrict__ xqp) {
    int gid = blockIdx.x * 256 + threadIdx.x;
    if (gid >= NB * 228 * 8) return;
    int c16 = gid & 7;
    int r   = (gid >> 3) % 228;
    int n   = (gid >> 3) / 228;
    int y, x;
    if      (r < 58)  { y = 0;           x = r;       }
    else if (r < 116) { y = HP - 1;      x = r - 58;  }
    else if (r < 172) { y = r - 116 + 1; x = 0;       }
    else              { y = r - 172 + 1; x = HP - 1;  }
    uint4 z; z.x = z.y = z.z = z.w = 0u;
    xqp[((size_t)(n * HP + y) * HP + x) * 8 + c16] = z;
}

__global__ __launch_bounds__(256)
void k_quant_x(const float* __restrict__ x, uint32_t* __restrict__ xqp) {
    __shared__ float tile[CIN * 33];
    const int blk = blockIdx.x;
    const int n   = blk / (HW / 32);
    const int hw0 = (blk - n * (HW / 32)) * 32;

    {
        const int pix = threadIdx.x & 31;
        const int c0  = threadIdx.x >> 5;
        const float* xb = x + (size_t)n * CIN * HW + hw0;
#pragma unroll
        for (int i = 0; i < 16; ++i) {
            int c = c0 + i * 8;
            tile[c * 33 + pix] = xb[(size_t)c * HW + pix];
        }
    }
    __syncthreads();
    {
        const int c4 = threadIdx.x & 31;
        const int p0 = threadIdx.x >> 5;
#pragma unroll
        for (int i = 0; i < 4; ++i) {
            int p  = p0 + i * 8;
            int hw = hw0 + p;
            int y  = hw / WW, xc = hw - y * WW;
            uint32_t packed = 0;
#pragma unroll
            for (int k = 0; k < 4; ++k) {
                float v = tile[(c4 * 4 + k) * 33 + p];
                v = fminf(fmaxf(v, 0.0f), CLIP_MAX);
                int q = (int)rintf(v * QA_SCALE);
                packed |= ((uint32_t)(q & 255)) << (8 * k);
            }
            xqp[((size_t)(n * HP + y + 1) * HP + (xc + 1)) * 32 + c4] = packed;
        }
    }
}

#define ROW_DW  36
#define A_DW    (64 * ROW_DW)
#define B_DW    (128 * ROW_DW)
#define BUF_DW  (A_DW + B_DW)

__global__ __launch_bounds__(256)
void k_conv_wmma(const uint32_t* __restrict__ xqp,
                 const uint32_t* __restrict__ wq,
                 const unsigned* __restrict__ smax,
                 const float* __restrict__ gamma, const float* __restrict__ beta,
                 const float* __restrict__ mean,  const float* __restrict__ var,
                 float* __restrict__ out) {
    __shared__ __align__(16) uint32_t smem[2 * BUF_DW];

    const int tid = threadIdx.x;
    const int lane = tid & 31;
    const int wv   = tid >> 5;
    const int wp   = wv & 3;
    const int wc   = wv >> 2;
    const int lh   = lane >> 4;
    const int l15  = lane & 15;

    const int pix_base  = blockIdx.x * 64;
    const int cout_base = blockIdx.y * 128;
    const int n   = pix_base / HW;
    const int hwb = pix_base - n * HW;

    const uint32_t* aSrc[2];
    int aLds[2];
#pragma unroll
    for (int i = 0; i < 2; ++i) {
        int d   = tid + i * 256;
        int p   = d >> 3;
        int c16 = d & 7;
        int hw  = hwb + p;
        int oy  = hw / WW, ox = hw - oy * WW;
        aSrc[i] = xqp + ((size_t)(n * HP + oy + 1) * HP + (ox + 1)) * 32 + c16 * 4;
        aLds[i] = p * ROW_DW + c16 * 4;
    }
    int bLds[4];
#pragma unroll
    for (int i = 0; i < 4; ++i) {
        int d = tid + i * 256;
        bLds[i] = A_DW + (d >> 3) * ROW_DW + (d & 7) * 4;
    }
    const uint32_t* bBase = wq + (size_t)cout_base * 32;

    auto issue_tap = [&](int t) {
        const int dy = t / 3 - 1;
        const int dx = t - 3 * (t / 3) - 1;
        const int aoff = (dy * HP + dx) * 32;
        uint32_t* buf = smem + (t & 1) * BUF_DW;
#pragma unroll
        for (int i = 0; i < 2; ++i)
            async_copy16(aSrc[i] + aoff, buf + aLds[i]);
        const uint32_t* bs = bBase + (size_t)t * (COUT * CIN / 4);
#pragma unroll
        for (int i = 0; i < 4; ++i)
            async_copy16(bs + (size_t)(tid + i * 256) * 4, buf + bLds[i]);
    };

    v8i acc[4] = {};
    const int aRow = (wp * 16 + l15) * ROW_DW;

    issue_tap(0);
    wait_async();
    __syncthreads();

    for (int t = 0; t < NTAPS; ++t) {
        if (t < NTAPS - 1) issue_tap(t + 1);

        const uint32_t* bufA = smem + (t & 1) * BUF_DW;
        const uint32_t* bufB = bufA + A_DW;
#pragma unroll
        for (int h2 = 0; h2 < 2; ++h2) {
            v8i a;
            const uint2* ap = reinterpret_cast<const uint2*>(bufA + aRow);
#pragma unroll
            for (int g = 0; g < 4; ++g) {
                uint2 t2 = ap[h2 * 8 + (g >> 1) * 4 + lh * 2 + (g & 1)];
                a[2 * g]     = (int)t2.x;
                a[2 * g + 1] = (int)t2.y;
            }
#pragma unroll
            for (int j = 0; j < 4; ++j) {
                const uint4* bp = reinterpret_cast<const uint4*>(
                    bufB + (wc * 64 + j * 16 + l15) * ROW_DW);
                uint4 b0 = bp[h2 * 4 + lh];
                uint4 b1 = bp[h2 * 4 + 2 + lh];
                v8i b;
                b[0] = (int)b0.x; b[1] = (int)b0.y; b[2] = (int)b0.z; b[3] = (int)b0.w;
                b[4] = (int)b1.x; b[5] = (int)b1.y; b[6] = (int)b1.z; b[7] = (int)b1.w;
                acc[j] = __builtin_amdgcn_wmma_i32_16x16x64_iu8(
                     false, a,  true, b, acc[j],
                     false,  false);
            }
        }
        wait_async();
        __syncthreads();
    }

    const float s  = fmaxf(__uint_as_float(*smax), 1e-8f);
    const float ab = QA_INV * (s / 127.0f);
    float* fsm = reinterpret_cast<float*>(smem);
#pragma unroll
    for (int j = 0; j < 4; ++j) {
        const int col = wc * 64 + j * 16 + l15;
        const int co  = cout_base + col;
        const float sc = gamma[co] * rsqrtf(var[co] + BN_EPS);
        const float mn = mean[co];
        const float bt = beta[co];
#pragma unroll
        for (int r = 0; r < 8; ++r) {
            int pl = wp * 16 + r + lh * 8;
            float y = (float)acc[j][r] * ab;
            y = (y - mn) * sc + bt;
            float yc = fminf(fmaxf(y, 0.0f), CLIP_MAX);
            fsm[col * 68 + pl] = rintf(yc * QA_SCALE) * QA_INV;
        }
    }
    __syncthreads();

    float* ob = out + ((size_t)n * COUT + cout_base) * HW + hwb;
#pragma unroll
    for (int i = 0; i < 8; ++i) {
        int idx = tid + i * 256;
        int row = idx >> 4;
        int c4  = idx & 15;
        uint4 v = *reinterpret_cast<const uint4*>(&smem[row * 68 + c4 * 4]);
        *reinterpret_cast<uint4*>(ob + (size_t)row * HW + c4 * 4) = v;
    }
}

extern "C" void kernel_launch(void* const* d_in, const int* in_sizes, int n_in,
                              void* d_out, int out_size, void* d_ws, size_t ws_size,
                              hipStream_t stream) {
    (void)in_sizes; (void)n_in; (void)out_size; (void)ws_size;
    const float* x      = (const float*)d_in[0];
    const float* weight = (const float*)d_in[1];
    const float* gamma  = (const float*)d_in[2];
    const float* beta   = (const float*)d_in[3];
    const float* mean   = (const float*)d_in[4];
    const float* var    = (const float*)d_in[5];
    float* out          = (float*)d_out;

    const size_t XQP_BYTES = (size_t)NB * HP * HP * CIN;
    unsigned*    smax = (unsigned*)d_ws;
    uint32_t*    xqp  = (uint32_t*)((char*)d_ws + 256);
    signed char* wq   = (signed char*)((char*)d_ws + 256 + XQP_BYTES);

    const int n_w = COUT * CIN * 9;

    k_init_smax<<<1, 64, 0, stream>>>(smax);
    k_absmax<<<256, 256, 0, stream>>>(weight, smax, n_w);
    k_quant_w<<<(n_w + 255) / 256, 256, 0, stream>>>(weight, smax, wq);
    k_zero_halo<<<(NB * 228 * 8) / 256, 256, 0, stream>>>((uint4*)xqp);
    k_quant_x<<<NB * (HW / 32), 256, 0, stream>>>(x, xqp);

    dim3 grid(NPIX / 64, COUT / 128, 1);
    k_conv_wmma<<<grid, 256, 0, stream>>>(xqp, (const uint32_t*)wq, smax,
                                          gamma, beta, mean, var, out);
}
